// CrossViewAttention_67293547594265
// MI455X (gfx1250) — matrix operand order repaired and checked by reading, not compiled, not run on hardware
//
#include <hip/hip_runtime.h>

// ---------------------------------------------------------------------------
// CrossViewAttention for MI455X (gfx1250, wave32, WMMA 16x16x32 f16)
//
//  K1 k_inv3       : 3x3 intrinsic inverse (B=2)
//  K2 k_bnstats    : BN mean/rstd per channel over (B,H,W)
//  K3 k_embed_pre  : algebraic reduction of img/bev positional-embedding GEMMs
//  K4 k_wcvt x6    : weights f32 -> f16, pre-transposed to WT[n][k]
//  K5 k_kvtok_ln   : image tokens -> BN/ReLU/conv -> (+img_pe) -> LN -> f16
//  K6 k_qtok_ln    : bev tokens  -> (+bev_pe) -> LN -> f16
//  K7 k_gemm (x3)  : WMMA head projections (k / v / q)
//  K8 k_attn       : flash attention; WMMA scores + P@V + P@ones (row-sum)
//  K9 k_gemm       : O @ projW (f32 out)
//  K10 k_ln        : LayerNorm(pre) -> f32 residual + f16 activations
//  K11 k_gemm      : GELU(z @ w1) (N=256, f16 out)
//  K12 k_gemm      : h @ w2 (K=256, f32 out)
//  K13 k_final     : residual + LN(post) + transpose out
// ---------------------------------------------------------------------------

typedef _Float16 half_t;
typedef __attribute__((ext_vector_type(16))) _Float16      v16h;
typedef __attribute__((ext_vector_type(8)))  float         v8f;
typedef __attribute__((ext_vector_type(4)))  unsigned int  u32x4;

#define BB      2
#define CIN     256
#define BEVD    128
#define FW      50
#define NKEY    1400          // 28*50
#define NKEYP   1408          // padded to 32
#define NQ      10000         // Z*X
#define XD      100
#define HEADS   4
#define DH      32

union V16 { u32x4 u[2]; v16h v; };

// ---------------- block reduction helper (power-of-2 block) ----------------
template<int N>
__device__ inline float blk_sum(float v, float* red) {
  int tid = threadIdx.x;
  red[tid] = v; __syncthreads();
  #pragma unroll
  for (int s = N / 2; s > 0; s >>= 1) {
    if (tid < s) red[tid] += red[tid + s];
    __syncthreads();
  }
  float r = red[0]; __syncthreads();
  return r;
}

// ---------------- K1: 3x3 inverse of scaled intrinsics ----------------------
__global__ void k_inv3(const float* __restrict__ intr, float* __restrict__ kinv) {
  int b = threadIdx.x;
  if (b >= BB) return;
  float m[9];
  #pragma unroll
  for (int i = 0; i < 3; i++) {
    float s = (i == 0) ? 0.125f : ((i == 1) ? 0.125f : 1.0f);  // SX, SY, 1
    #pragma unroll
    for (int j = 0; j < 3; j++) m[i * 3 + j] = intr[b * 9 + i * 3 + j] * s;
  }
  float c00 =  m[4] * m[8] - m[5] * m[7];
  float c01 = -(m[3] * m[8] - m[5] * m[6]);
  float c02 =  m[3] * m[7] - m[4] * m[6];
  float id  = 1.0f / (m[0] * c00 + m[1] * c01 + m[2] * c02);
  float inv[9];
  inv[0] = c00 * id;                              inv[3] = c01 * id;
  inv[6] = c02 * id;
  inv[1] = -(m[1] * m[8] - m[2] * m[7]) * id;     inv[4] =  (m[0] * m[8] - m[2] * m[6]) * id;
  inv[7] = -(m[0] * m[7] - m[1] * m[6]) * id;
  inv[2] =  (m[1] * m[5] - m[2] * m[4]) * id;     inv[5] = -(m[0] * m[5] - m[2] * m[3]) * id;
  inv[8] =  (m[0] * m[4] - m[1] * m[3]) * id;
  #pragma unroll
  for (int i = 0; i < 9; i++) kinv[b * 9 + i] = inv[i];
}

// ---------------- K2: BatchNorm statistics over (B,H,W) ---------------------
__global__ void __launch_bounds__(256) k_bnstats(const float* __restrict__ img,
                                                 float* __restrict__ mean,
                                                 float* __restrict__ rstd) {
  __shared__ float red[256];
  int c = blockIdx.x;
  float s = 0.f, s2 = 0.f;
  for (int i = threadIdx.x; i < BB * NKEY; i += 256) {
    int b = i / NKEY, hw = i % NKEY;
    float x = img[((size_t)(b * CIN + c)) * NKEY + hw];
    s += x; s2 += x * x;
  }
  float ts  = blk_sum<256>(s,  red);
  float ts2 = blk_sum<256>(s2, red);
  if (threadIdx.x == 0) {
    float m = ts / (BB * NKEY);
    mean[c] = m;
    rstd[c] = rsqrtf(ts2 / (BB * NKEY) - m * m + 1e-5f);
  }
}

// ---------------- K3: positional-embedding weight reductions ----------------
__global__ void k_embed_pre(const float* __restrict__ imgW,
                            const float* __restrict__ bevW,
                            const float* __restrict__ bevB,
                            float* __restrict__ imgS, float* __restrict__ imgT,
                            float* __restrict__ S0, float* __restrict__ S2,
                            float* __restrict__ C1b) {
  int o = threadIdx.x;   // 0..127
  #pragma unroll
  for (int c3 = 0; c3 < 3; c3++) {
    float s = 0.f, t = 0.f;
    for (int d = 0; d < 64; d++) {
      float w = imgW[(c3 * 64 + d) * BEVD + o];
      s += w;
      t += (d * (1.0f / 63.0f)) * 64.0f * w;   // pz(d)
    }
    imgS[c3 * BEVD + o] = s;
    imgT[c3 * BEVD + o] = t;
  }
  float s0 = 0.f, s2v = 0.f, c1 = bevB[o];
  for (int y = 0; y < 8; y++) {
    s0  += bevW[(0 * 8 + y) * BEVD + o];
    s2v += bevW[(16 + y) * BEVD + o];
    c1  += ((y * (1.0f / 7.0f)) - 0.5f) * 8.0f * bevW[(8 + y) * BEVD + o];
  }
  S0[o] = s0; S2[o] = s2v; C1b[o] = c1;
}

// ---------------- K4: weight convert + transpose (W[k][n] -> WT[n][k] f16) --
__global__ void __launch_bounds__(256) k_wcvt(const float* __restrict__ W,
                                              half_t* __restrict__ WT,
                                              int K, int N) {
  int i = blockIdx.x * 256 + threadIdx.x;
  if (i >= K * N) return;
  int k = i / N, n = i % N;
  WT[(size_t)n * K + k] = (half_t)W[i];
}

// ---------------- K5: image-side tokens, up to LayerNorm (f16 out) ----------
// attention K comes from the val path, attention V from key path + img_pe
__global__ void __launch_bounds__(128) k_kvtok_ln(
    const float* __restrict__ img,  const float* __restrict__ kinv,
    const float* __restrict__ mean, const float* __restrict__ rstd,
    const float* __restrict__ kbg,  const float* __restrict__ kbb,
    const float* __restrict__ kcw,
    const float* __restrict__ vbg,  const float* __restrict__ vbb,
    const float* __restrict__ vcw,
    const float* __restrict__ imgS, const float* __restrict__ imgT,
    const float* __restrict__ img_eb,
    const float* __restrict__ lnkg, const float* __restrict__ lnkb,
    const float* __restrict__ lnvg, const float* __restrict__ lnvb,
    half_t* __restrict__ lnKh, half_t* __restrict__ lnVh) {
  __shared__ float fK[CIN], fV[CIN], red[BEVD];
  int t = blockIdx.x % NKEYP;
  int b = blockIdx.x / NKEYP;
  int o = threadIdx.x;
  size_t rowbase = ((size_t)(b * NKEYP + t)) * BEVD;
  if (t >= NKEY) {                 // zero padded key tokens (A rows of GEMM)
    lnKh[rowbase + o] = (half_t)0.f;
    lnVh[rowbase + o] = (half_t)0.f;
    return;
  }
  #pragma unroll
  for (int cc = 0; cc < 2; cc++) {
    int c = o + cc * 128;
    float x  = img[((size_t)(b * CIN + c)) * NKEY + t];
    float xn = (x - mean[c]) * rstd[c];
    fK[c] = fmaxf(0.f, xn * kbg[c] + kbb[c]);
    fV[c] = fmaxf(0.f, xn * vbg[c] + vbb[c]);
  }
  __syncthreads();
  float aK = 0.f, aV = 0.f;
  for (int c = 0; c < CIN; c++) {
    aK += fV[c] * vcw[o * CIN + c];   // value path -> attention K
    aV += fK[c] * kcw[o * CIN + c];   // key   path -> attention V
  }
  int hh = t / FW, wwi = t % FW;
  float px = (wwi * (1.0f / 49.0f)) * 3200.0f;
  float py = (hh  * (1.0f / 27.0f)) * 1792.0f;
  float pe = img_eb[o];
  #pragma unroll
  for (int c3 = 0; c3 < 3; c3++) {
    pe += (kinv[b * 9 + c3 * 3 + 0] * px + kinv[b * 9 + c3 * 3 + 1] * py) * imgS[c3 * BEVD + o]
        +  kinv[b * 9 + c3 * 3 + 2] * imgT[c3 * BEVD + o];
  }
  float rawK = aK;
  float rawV = pe + aV;
  float sK  = blk_sum<128>(rawK, red);
  float sK2 = blk_sum<128>(rawK * rawK, red);
  float sV  = blk_sum<128>(rawV, red);
  float sV2 = blk_sum<128>(rawV * rawV, red);
  float mK = sK * (1.f / 128.f), vK = sK2 * (1.f / 128.f) - mK * mK;
  float mV = sV * (1.f / 128.f), vV = sV2 * (1.f / 128.f) - mV * mV;
  lnKh[rowbase + o] = (half_t)((rawK - mK) * rsqrtf(vK + 1e-5f) * lnkg[o] + lnkb[o]);
  lnVh[rowbase + o] = (half_t)((rawV - mV) * rsqrtf(vV + 1e-5f) * lnvg[o] + lnvb[o]);
}

// ---------------- K6: query tokens, up to LayerNorm (f16 out) ---------------
__global__ void __launch_bounds__(128) k_qtok_ln(
    const float* __restrict__ bev,
    const float* __restrict__ S0, const float* __restrict__ S2,
    const float* __restrict__ C1b,
    const float* __restrict__ lnqg, const float* __restrict__ lnqb,
    half_t* __restrict__ Qlnh) {
  __shared__ float red[BEVD];
  int t = blockIdx.x % NQ;
  int b = blockIdx.x / NQ;
  int o = threadIdx.x;
  int zi = t / XD, xi = t % XD;
  float px = (xi * (1.0f / 99.0f) - 0.5f) * 100.0f;
  float pz = (zi * (1.0f / 99.0f)) * 100.0f;
  float raw = bev[((size_t)(b * BEVD + o)) * NQ + t]
            + px * S0[o] + pz * S2[o] + C1b[o];
  float s  = blk_sum<128>(raw, red);
  float s2 = blk_sum<128>(raw * raw, red);
  float mm = s * (1.f / 128.f), var = s2 * (1.f / 128.f) - mm * mm;
  Qlnh[((size_t)(b * NQ + t)) * BEVD + o] =
      (half_t)((raw - mm) * rsqrtf(var + 1e-5f) * lnqg[o] + lnqb[o]);
}

// ---------------- K7/K9/K11/K12: WMMA GEMM  out = act(A @ WT^T + bias) ------
// A: M x KD f16 (row-major); WT: N x KD f16 (pre-transposed weights).
// One wave computes a 16-row tile; B fragments gather straight from global
// (weights are L2-resident). ACT: 0=none, 1=exact GELU. OUT16: f16/f32 out.
template<int N, int KD, int ACT, int OUT16>
__global__ void __launch_bounds__(128) k_gemm(
    const half_t* __restrict__ A, const half_t* __restrict__ WT,
    const float* __restrict__ bias, void* __restrict__ outp, int M) {
  const int wave = threadIdx.x >> 5, lane = threadIdx.x & 31;
  const int g = lane >> 4, ln = lane & 15, kh = g * 8;
  const int m0 = blockIdx.x * 64 + wave * 16;
  if (m0 >= M) return;                     // wave-uniform; kernel has no barriers
  constexpr int NT = N / 16;
  v8f acc[NT];
  #pragma unroll
  for (int t = 0; t < NT; t++) acc[t] = {};
  #pragma unroll
  for (int ks = 0; ks < KD; ks += 32) {
    V16 a;                                 // A fragment: 16 rows x 32 k
    const half_t* ar = A + ((size_t)(m0 + ln)) * KD + ks;
    a.u[0] = *(const u32x4*)(ar + 16 * g);
    a.u[1] = *(const u32x4*)(ar + 16 * g + 8);
    #pragma unroll
    for (int t = 0; t < NT; t++) {
      V16 bf;                              // B fragment: rows k = ks+16g+e
      const half_t* wr = WT + ((size_t)(t * 16 + ln)) * KD + ks + 16 * g;
      bf.u[0] = *(const u32x4*)wr;
      bf.u[1] = *(const u32x4*)(wr + 8);
      acc[t] = __builtin_amdgcn_wmma_f32_16x16x32_f16(false, a.v, false, bf.v,
                                                      (short)0, acc[t], false, false);
    }
  }
  #pragma unroll
  for (int t = 0; t < NT; t++) {
    int n = t * 16 + ln;
    float bn = bias[n];
    #pragma unroll
    for (int r = 0; r < 8; r++) {
      size_t row = (size_t)(m0 + r + 8 * g);
      float v = acc[t][r] + bn;
      if (ACT == 1) v = 0.5f * v * (1.0f + erff(v * 0.70710678118654752f));
      if (OUT16) ((half_t*)outp)[row * N + n] = (half_t)v;
      else       ((float*) outp)[row * N + n] = v;
    }
  }
}

// ---------------- K8: flash attention, WMMA f16->f32 ------------------------
// block = 128 threads (4 waves); wave -> one 16-query tile; grid (157, B*H).
// Row-sum of the softmax numerator is computed on the matrix unit via an
// all-ones B fragment (every column of P@1 carries the row sum), replacing
// 32 ds_bpermute per slab.
__global__ void __launch_bounds__(128) k_attn(
    const half_t* __restrict__ Qh, const half_t* __restrict__ Kh,
    const half_t* __restrict__ Vh, half_t* __restrict__ Oh) {
  __shared__ __align__(16) half_t ldsK [32 * 32];   // [key][dh]
  __shared__ __align__(16) half_t ldsVT[32 * 32];   // [dh][key]
  __shared__ __align__(16) half_t Pbuf [4][16 * 32];

  const int bh   = blockIdx.y;
  const int b    = bh >> 2;
  const int head = bh & 3;
  const int wave = threadIdx.x >> 5;
  const int lane = threadIdx.x & 31;
  const int g    = lane >> 4;
  const int ln   = lane & 15;
  const int kh   = g * 8;
  const int q0   = blockIdx.x * 64 + wave * 16;
  const bool active = (q0 < NQ);     // wave-uniform

  V16 qa;
  { u32x4 z = {0u, 0u, 0u, 0u}; qa.u[0] = z; qa.u[1] = z; }
  if (active) {
    const half_t* qrow = Qh + ((size_t)(b * NQ + q0 + ln)) * BEVD + head * DH;
    qa.u[0] = *(const u32x4*)(qrow + 16 * g);
    qa.u[1] = *(const u32x4*)(qrow + 16 * g + 8);
  }
  V16 onesf;
  #pragma unroll
  for (int i = 0; i < 16; i++) onesf.v[i] = (half_t)1.0f;

  v8f o0 = {}; v8f o1 = {}; v8f lacc = {};
  float m[8];
  #pragma unroll
  for (int r = 0; r < 8; r++) m[r] = -3.0e38f;
  const float scale = 0.17677669529663687f;   // 1/sqrt(DH)

  for (int slab = 0; slab < NKEYP; slab += 32) {
    {
      int key = threadIdx.x >> 2;
      int dh8 = (threadIdx.x & 3) * 8;
      size_t gidx = ((size_t)(b * NKEYP + slab + key)) * BEVD + head * DH + dh8;
      u32x4 kv = *(const u32x4*)(Kh + gidx);
      *(u32x4*)&ldsK[key * 32 + dh8] = kv;
      u32x4 vv = *(const u32x4*)(Vh + gidx);
      const half_t* hv = (const half_t*)&vv;
      #pragma unroll
      for (int j = 0; j < 8; j++) ldsVT[(dh8 + j) * 32 + key] = hv[j];
      if (slab + 32 < NKEYP) {                       // gfx1250 global_prefetch_b8
        __builtin_prefetch(Kh + gidx + 32 * BEVD, 0, 1);
        __builtin_prefetch(Vh + gidx + 32 * BEVD, 0, 1);
      }
    }
    __syncthreads();

    V16 kb0, kb1;
    {
      const half_t* kp0 = &ldsK[ln * 32 + g * 16];
      kb0.u[0] = *(const u32x4*)kp0;       kb0.u[1] = *(const u32x4*)(kp0 + 8);
      const half_t* kp1 = &ldsK[(ln + 16) * 32 + g * 16];
      kb1.u[0] = *(const u32x4*)kp1;       kb1.u[1] = *(const u32x4*)(kp1 + 8);
    }
    v8f zc = {};
    v8f s0 = __builtin_amdgcn_wmma_f32_16x16x32_f16(false, qa.v, false, kb0.v,
                                                    (short)0, zc, false, false);
    v8f s1 = __builtin_amdgcn_wmma_f32_16x16x32_f16(false, qa.v, false, kb1.v,
                                                    (short)0, zc, false, false);

    const bool valid0 = (slab + ln)      < NKEY;
    const bool valid1 = (slab + 16 + ln) < NKEY;

    float p0[8], p1[8];
    #pragma unroll
    for (int r = 0; r < 8; r++) {
      float a  = valid0 ? s0[r] * scale : -3.0e38f;
      float c  = valid1 ? s1[r] * scale : -3.0e38f;
      float v  = fmaxf(a, c);
      v = fmaxf(v, __shfl_xor(v, 1, 32));
      v = fmaxf(v, __shfl_xor(v, 2, 32));
      v = fmaxf(v, __shfl_xor(v, 4, 32));
      v = fmaxf(v, __shfl_xor(v, 8, 32));
      float mn    = fmaxf(m[r], v);
      float alpha = __expf(m[r] - mn);
      m[r] = mn;
      p0[r] = valid0 ? __expf(s0[r] * scale - mn) : 0.f;
      p1[r] = valid1 ? __expf(s1[r] * scale - mn) : 0.f;
      lacc[r] *= alpha;                    // rescale running denominator
      o0[r]   *= alpha;
      o1[r]   *= alpha;
    }

    half_t* pb = Pbuf[wave];
    #pragma unroll
    for (int r = 0; r < 8; r++) {
      pb[(r + 8 * g) * 32 + ln]      = (half_t)p0[r];
      pb[(r + 8 * g) * 32 + ln + 16] = (half_t)p1[r];
    }
    V16 pa;
    {
      const half_t* pr = &pb[ln * 32];     // same-wave LDS ops are in-order
      pa.u[0] = *(const u32x4*)(pr + 16 * g);
      pa.u[1] = *(const u32x4*)(pr + 16 * g + 8);
    }
    V16 vb0, vb1;
    {
      const half_t* vp0 = &ldsVT[ln * 32 + 16 * g];
      vb0.u[0] = *(const u32x4*)vp0;       vb0.u[1] = *(const u32x4*)(vp0 + 8);
      const half_t* vp1 = &ldsVT[(ln + 16) * 32 + 16 * g];
      vb1.u[0] = *(const u32x4*)vp1;       vb1.u[1] = *(const u32x4*)(vp1 + 8);
    }
    o0 = __builtin_amdgcn_wmma_f32_16x16x32_f16(false, pa.v, false, vb0.v,
                                                (short)0, o0, false, false);
    o1 = __builtin_amdgcn_wmma_f32_16x16x32_f16(false, pa.v, false, vb1.v,
                                                (short)0, o1, false, false);
    lacc = __builtin_amdgcn_wmma_f32_16x16x32_f16(false, pa.v, false, onesf.v,
                                                  (short)0, lacc, false, false);
    __syncthreads();
  }

  if (active) {
    #pragma unroll
    for (int r = 0; r < 8; r++) {
      float inv = 1.0f / lacc[r];
      size_t row = ((size_t)(b * NQ + q0 + r + 8 * g)) * BEVD + head * DH;
      Oh[row + ln]      = (half_t)(o0[r] * inv);
      Oh[row + ln + 16] = (half_t)(o1[r] * inv);
    }
  }
}

// ---------------- K10: LayerNorm -> f32 residual + f16 activations ----------
__global__ void __launch_bounds__(128) k_ln(const float* __restrict__ in,
                                            const float* __restrict__ gg,
                                            const float* __restrict__ bb,
                                            float* __restrict__ outf,
                                            half_t* __restrict__ outh) {
  __shared__ float red[BEVD];
  size_t t = blockIdx.x;
  int o = threadIdx.x;
  float z = in[t * BEVD + o];
  float s  = blk_sum<128>(z, red);
  float s2 = blk_sum<128>(z * z, red);
  float mm = s * (1.f / 128.f), var = s2 * (1.f / 128.f) - mm * mm;
  float zl = (z - mm) * rsqrtf(var + 1e-5f) * gg[o] + bb[o];
  outf[t * BEVD + o] = zl;
  outh[t * BEVD + o] = (half_t)zl;
}

// ---------------- K13: residual + LN(post) + transpose ----------------------
__global__ void __launch_bounds__(128) k_final(const float* __restrict__ Zl,
                                               const float* __restrict__ R,
                                               const float* __restrict__ postg,
                                               const float* __restrict__ postb,
                                               float* __restrict__ out) {
  __shared__ float red[BEVD];
  int t = blockIdx.x % NQ;
  int b = blockIdx.x / NQ;
  int o = threadIdx.x;
  size_t tok = (size_t)blockIdx.x;
  float y = Zl[tok * BEVD + o] + R[tok * BEVD + o];
  float s  = blk_sum<128>(y, red);
  float s2 = blk_sum<128>(y * y, red);
  float mm = s * (1.f / 128.f), var = s2 * (1.f / 128.f) - mm * mm;
  out[((size_t)(b * BEVD + o)) * NQ + t] =
      (y - mm) * rsqrtf(var + 1e-5f) * postg[o] + postb[o];
}

// ---------------------------------------------------------------------------
extern "C" void kernel_launch(void* const* d_in, const int* in_sizes, int n_in,
                              void* d_out, int out_size, void* d_ws, size_t ws_size,
                              hipStream_t stream) {
  const float* img    = (const float*)d_in[0];
  const float* intr   = (const float*)d_in[1];
  const float* bev    = (const float*)d_in[2];
  const float* imgEw  = (const float*)d_in[3];
  const float* imgEb  = (const float*)d_in[4];
  const float* bevEw  = (const float*)d_in[5];
  const float* bevEb  = (const float*)d_in[6];
  const float* kbng   = (const float*)d_in[7];
  const float* kbnb   = (const float*)d_in[8];
  const float* kconv  = (const float*)d_in[9];
  const float* vbng   = (const float*)d_in[10];
  const float* vbnb   = (const float*)d_in[11];
  const float* vconv  = (const float*)d_in[12];
  const float* lnqg   = (const float*)d_in[13];
  const float* lnqb   = (const float*)d_in[14];
  const float* lnkg   = (const float*)d_in[15];
  const float* lnkb   = (const float*)d_in[16];
  const float* lnvg   = (const float*)d_in[17];
  const float* lnvb   = (const float*)d_in[18];
  const float* wq     = (const float*)d_in[19];
  const float* bq     = (const float*)d_in[20];
  const float* wk     = (const float*)d_in[21];
  const float* bk     = (const float*)d_in[22];
  const float* wv     = (const float*)d_in[23];
  const float* bv     = (const float*)d_in[24];
  const float* projW  = (const float*)d_in[25];
  const float* projB  = (const float*)d_in[26];
  const float* preg   = (const float*)d_in[27];
  const float* preb   = (const float*)d_in[28];
  const float* w1     = (const float*)d_in[29];
  const float* b1     = (const float*)d_in[30];
  const float* w2     = (const float*)d_in[31];
  const float* b2     = (const float*)d_in[32];
  const float* postg  = (const float*)d_in[33];
  const float* postb  = (const float*)d_in[34];
  float* out = (float*)d_out;

  // ---- workspace layout -------------------------------------------------
  float* wsf     = (float*)d_ws;
  float* kinv    = wsf + 0;       // 18
  float* bnMean  = wsf + 32;      // 256
  float* bnRstd  = wsf + 288;     // 256
  float* imgS    = wsf + 544;     // 384
  float* imgT    = wsf + 928;     // 384
  float* bS0     = wsf + 1312;    // 128
  float* bS2     = wsf + 1440;    // 128
  float* bC1     = wsf + 1568;    // 128
  float* Zl      = wsf + 1728;                      // 2.56M f32 (LN'd residual)
  float* Zbuf    = Zl + (size_t)BB * NQ * BEVD;     // 2.56M f32 (proj out, reused as R)
  float* Rbuf    = Zbuf;                            // alias: Zbuf dead after k_ln
  half_t* hbase  = (half_t*)(Zbuf + (size_t)BB * NQ * BEVD);
  half_t* wqT    = hbase;               // 16384 each
  half_t* wkT    = wqT   + 16384;
  half_t* wvT    = wkT   + 16384;
  half_t* projWT = wvT   + 16384;
  half_t* w1T    = projWT + 16384;      // 32768 (256x128)
  half_t* w2T    = w1T   + 32768;       // 32768 (128x256)
  half_t* lnKh   = w2T   + 32768;       // 2*1408*128
  half_t* lnVh   = lnKh  + (size_t)BB * NKEYP * BEVD;
  half_t* Kh     = lnVh  + (size_t)BB * NKEYP * BEVD;
  half_t* Vh     = Kh    + (size_t)BB * NKEYP * BEVD;
  half_t* Qlnh   = Vh    + (size_t)BB * NKEYP * BEVD;   // 2.56M
  half_t* Qh     = Qlnh  + (size_t)BB * NQ * BEVD;      // 2.56M
  half_t* Ohh    = Qh    + (size_t)BB * NQ * BEVD;      // 2.56M
  half_t* Zlh    = Qlnh;                // alias: Qln dead after q-proj GEMM
  half_t* Hh     = Qh;                  // alias: Qh+Oh dead after proj GEMM (5.12M)
  // total ws use ~39 MB

  const int MKV = BB * NKEYP;   // 2816
  const int MQ  = BB * NQ;      // 20000

  k_inv3<<<1, 32, 0, stream>>>(intr, kinv);
  k_bnstats<<<CIN, 256, 0, stream>>>(img, bnMean, bnRstd);
  k_embed_pre<<<1, BEVD, 0, stream>>>(imgEw, bevEw, bevEb, imgS, imgT, bS0, bS2, bC1);

  k_wcvt<<<64, 256, 0, stream>>>(wq,    wqT,    128, 128);
  k_wcvt<<<64, 256, 0, stream>>>(wk,    wkT,    128, 128);
  k_wcvt<<<64, 256, 0, stream>>>(wv,    wvT,    128, 128);
  k_wcvt<<<64, 256, 0, stream>>>(projW, projWT, 128, 128);
  k_wcvt<<<128, 256, 0, stream>>>(w1,   w1T,    128, 256);
  k_wcvt<<<128, 256, 0, stream>>>(w2,   w2T,    256, 128);

  k_kvtok_ln<<<MKV, 128, 0, stream>>>(img, kinv, bnMean, bnRstd,
                                      kbng, kbnb, kconv, vbng, vbnb, vconv,
                                      imgS, imgT, imgEb,
                                      lnkg, lnkb, lnvg, lnvb, lnKh, lnVh);
  k_gemm<128, 128, 0, 1><<<MKV / 64, 128, 0, stream>>>(lnKh, wkT, bk, Kh, MKV);
  k_gemm<128, 128, 0, 1><<<MKV / 64, 128, 0, stream>>>(lnVh, wvT, bv, Vh, MKV);

  k_qtok_ln<<<MQ, 128, 0, stream>>>(bev, bS0, bS2, bC1, lnqg, lnqb, Qlnh);
  k_gemm<128, 128, 0, 1><<<(MQ + 63) / 64, 128, 0, stream>>>(Qlnh, wqT, bq, Qh, MQ);

  dim3 agrid((NQ + 63) / 64, BB * HEADS);
  k_attn<<<agrid, 128, 0, stream>>>(Qh, Kh, Vh, Ohh);

  k_gemm<128, 128, 0, 0><<<(MQ + 63) / 64, 128, 0, stream>>>(Ohh, projWT, projB, Zbuf, MQ);
  k_ln<<<MQ, 128, 0, stream>>>(Zbuf, preg, preb, Zl, Zlh);
  k_gemm<256, 128, 1, 1><<<(MQ + 63) / 64, 128, 0, stream>>>(Zlh, w1T, b1, Hh, MQ);
  k_gemm<128, 256, 0, 0><<<(MQ + 63) / 64, 128, 0, stream>>>(Hh, w2T, b2, Rbuf, MQ);
  k_final<<<MQ, 128, 0, stream>>>(Zl, Rbuf, postg, postb, out);
}
